// LST_9148280340976
// MI455X (gfx1250) — matrix operand order repaired and checked by reading, not compiled, not run on hardware
//
#include <hip/hip_runtime.h>
#include <math.h>

typedef __attribute__((ext_vector_type(16))) _Float16 v16h;
typedef __attribute__((ext_vector_type(8)))  float    v8f;

#define NEG_INF_F (-1.0e9f)

// ---------------------------------------------------------------------------
// WMMA helpers (CDNA5 gfx1250, wave32): D = A(16x32 f16) * B(32x16 f16) + C(f32)
// ---------------------------------------------------------------------------
__device__ __forceinline__ v8f wmma_f16(v16h a, v16h b, v8f c) {
  return __builtin_amdgcn_wmma_f32_16x16x32_f16(false, a, false, b, (short)0, c,
                                                false, false);
}

__device__ __forceinline__ float wave_sum(float v) {
#pragma unroll
  for (int o = 16; o > 0; o >>= 1) v += __shfl_xor(v, o, 32);
  return v;
}

// A-fragment from row-major f16 (M x K), same contiguous k order as the B helpers:
// lane m (0-15)/(16-31): row m, K 0..15 / 16..31 (contiguous)
__device__ __forceinline__ v16h a_frag(const _Float16* A, int ld, int m0,
                                       int k0, int lane) {
  const _Float16* p = A + (size_t)(m0 + (lane & 15)) * ld + k0;
  int koff = (lane >> 4) * 16;
  v16h r;
#pragma unroll
  for (int e = 0; e < 8; e++) r[e] = p[koff + e];
#pragma unroll
  for (int e = 0; e < 8; e++) r[8 + e] = p[koff + 8 + e];
  return r;
}

// B-fragment where B = X^T and X is row-major (N x K):
// lane n holds B[k][n] = X[n][k], k contiguous per lane half.
__device__ __forceinline__ v16h bT_frag(const _Float16* X, int ld, int n0,
                                        int k0, int lane) {
  const _Float16* p =
      X + (size_t)(n0 + (lane & 15)) * ld + k0 + (lane >> 4) * 16;
  v16h r;
#pragma unroll
  for (int e = 0; e < 16; e++) r[e] = p[e];
  return r;
}

// B-fragment from row-major B (K x N): lane n reads column n, strided by ld.
__device__ __forceinline__ v16h b_frag(const _Float16* B, int ld, int k0,
                                       int n0, int lane) {
  int k = k0 + (lane >> 4) * 16;
  int n = n0 + (lane & 15);
  v16h r;
#pragma unroll
  for (int e = 0; e < 16; e++) r[e] = B[(size_t)(k + e) * ld + n];
  return r;
}

// ---------------------------------------------------------------------------
// Weight / activation packing into fragment-native layouts
// packed tile = 512 f16 (32 lanes x 16 halves), tile index lexicographic.
// A-packed: tile (mt,kt) index = mt*KT+kt.  B-packed: tile (kt,nt) index = nt*KT+kt.
// ---------------------------------------------------------------------------
__global__ __launch_bounds__(128) void pack_a_kernel(
    const float* __restrict__ X, _Float16* __restrict__ out, int KT, int K) {
  int wid = (blockIdx.x * 128 + threadIdx.x) >> 5;
  int lane = threadIdx.x & 31;
  int mt = wid / KT, kt = wid % KT;
  const float* p = X + (size_t)(mt * 16 + (lane & 15)) * K + kt * 32;
  int koff = (lane >> 4) * 16;
  _Float16* o = out + ((size_t)wid * 32 + lane) * 16;
#pragma unroll
  for (int e = 0; e < 8; e++) o[e] = (_Float16)p[koff + e];
#pragma unroll
  for (int e = 0; e < 8; e++) o[8 + e] = (_Float16)p[koff + 8 + e];
}

__global__ __launch_bounds__(128) void pack_b_kernel(
    const float* __restrict__ W, _Float16* __restrict__ out, int KT, int NW,
    int ntoff) {
  int wid = (blockIdx.x * 128 + threadIdx.x) >> 5;
  int lane = threadIdx.x & 31;
  int kt = wid % KT;
  int ntl = wid / KT;
  int col = ntl * 16 + (lane & 15);
  int k0 = kt * 32 + (lane >> 4) * 16;
  _Float16* o = out + (((size_t)(ntoff + ntl) * KT + kt) * 32 + lane) * 16;
#pragma unroll
  for (int e = 0; e < 16; e++) o[e] = (_Float16)W[(size_t)(k0 + e) * NW + col];
}

__global__ void concat_bias_kernel(const float* bq, const float* bk,
                                   const float* bv, float* out) {
  const float* src = (blockIdx.x == 0) ? bq : (blockIdx.x == 1) ? bk : bv;
  out[blockIdx.x * 256 + threadIdx.x] = src[threadIdx.x];
}

// ---------------------------------------------------------------------------
// Generic WMMA GEMM: C[M,N] = Apk[M,K] * Bpk[K,N] + bias (+relu) (+resid)
// one wave computes a 16(M) x 64(N) strip; grid is exact (all waves active).
// ---------------------------------------------------------------------------
template <int KT, bool RELU, bool RESID>
__global__ __launch_bounds__(128) void gemm_kernel(
    const _Float16* __restrict__ Apk, const _Float16* __restrict__ Bpk,
    const float* __restrict__ bias, const float* resid, float* C, int NT4,
    int N) {
  int wid = (blockIdx.x * 128 + threadIdx.x) >> 5;
  int lane = threadIdx.x & 31;
  int mt = wid / NT4, nt4 = wid % NT4;

  v8f acc[4];
  int ncol0 = nt4 * 64 + (lane & 15);
#pragma unroll
  for (int j = 0; j < 4; j++) {
    float bv = bias[ncol0 + j * 16];
#pragma unroll
    for (int v = 0; v < 8; v++) acc[j][v] = bv;
  }

  const _Float16* abase = Apk + (size_t)mt * KT * 512;
#pragma unroll 4
  for (int kt = 0; kt < KT; kt++) {
    v16h a = *(const v16h*)(abase + kt * 512 + lane * 16);
#pragma unroll
    for (int j = 0; j < 4; j++) {
      const _Float16* bb =
          Bpk + ((size_t)(nt4 * 4 + j) * KT + kt) * 512 + lane * 16;
      v16h b = *(const v16h*)bb;
      acc[j] = wmma_f16(a, b, acc[j]);
    }
  }

  int rbase = mt * 16 + 8 * (lane >> 4);
#pragma unroll
  for (int j = 0; j < 4; j++) {
    int col = nt4 * 64 + j * 16 + (lane & 15);
#pragma unroll
    for (int v = 0; v < 8; v++) {
      float x = acc[j][v];
      if (RELU) x = fmaxf(x, 0.0f);
      size_t idx = (size_t)(rbase + v) * N + col;
      if (RESID) x += resid[idx];
      C[idx] = x;
    }
  }
}

// ---------------------------------------------------------------------------
// Intra-session attention: one wave per session (12800 sessions).
// X = item_emb[ids] (20x256, zero rows for id==0); scores = XX^T/16;
// masked softmax over keys; E_hat = sum_q P X = (col-sums of P) . X
// ---------------------------------------------------------------------------
__global__ __launch_bounds__(32) void intra_kernel(
    const float* __restrict__ item_emb, const int* __restrict__ sessions,
    float* __restrict__ Ehat) {
  __shared__ _Float16 Xh[32 * 256];
  __shared__ float Sc[32 * 32];
  __shared__ float wcol[32];
  __shared__ int ids[32];

  int s = blockIdx.x;
  int lane = threadIdx.x;
  const int* sid = sessions + (size_t)s * 20;
  ids[lane] = (lane < 20) ? sid[lane] : 0;
  __syncthreads();

  // gather rows (8 f32 / lane / row), convert to f16 in LDS; pad rows 20..31
  for (int r = 0; r < 20; r++) {
    int id = ids[r];
    float4 f0 = make_float4(0.f, 0.f, 0.f, 0.f), f1 = f0;
    if (id != 0) {
      const float4* p = (const float4*)(item_emb + (size_t)id * 256 + lane * 8);
      f0 = p[0];
      f1 = p[1];
    }
    _Float16* d = Xh + r * 256 + lane * 8;
    d[0] = (_Float16)f0.x; d[1] = (_Float16)f0.y;
    d[2] = (_Float16)f0.z; d[3] = (_Float16)f0.w;
    d[4] = (_Float16)f1.x; d[5] = (_Float16)f1.y;
    d[6] = (_Float16)f1.z; d[7] = (_Float16)f1.w;
  }
  for (int r = 20; r < 32; r++) {
    _Float16* d = Xh + r * 256 + lane * 8;
#pragma unroll
    for (int t = 0; t < 8; t++) d[t] = (_Float16)0.0f;
  }
  __syncthreads();

  // scores = X X^T (32x32 padded), K = 256 = 8 k-tiles
  v8f acc[2][2];
#pragma unroll
  for (int mt = 0; mt < 2; mt++)
#pragma unroll
    for (int nt = 0; nt < 2; nt++)
#pragma unroll
      for (int v = 0; v < 8; v++) acc[mt][nt][v] = 0.0f;

  for (int kt = 0; kt < 8; kt++) {
    v16h a0 = a_frag(Xh, 256, 0, kt * 32, lane);
    v16h a1 = a_frag(Xh, 256, 16, kt * 32, lane);
    v16h b0 = bT_frag(Xh, 256, 0, kt * 32, lane);
    v16h b1 = bT_frag(Xh, 256, 16, kt * 32, lane);
    acc[0][0] = wmma_f16(a0, b0, acc[0][0]);
    acc[0][1] = wmma_f16(a0, b1, acc[0][1]);
    acc[1][0] = wmma_f16(a1, b0, acc[1][0]);
    acc[1][1] = wmma_f16(a1, b1, acc[1][1]);
  }
#pragma unroll
  for (int mt = 0; mt < 2; mt++)
#pragma unroll
    for (int nt = 0; nt < 2; nt++)
#pragma unroll
      for (int v = 0; v < 8; v++)
        Sc[(mt * 16 + v + 8 * (lane >> 4)) * 32 + nt * 16 + (lane & 15)] =
            acc[mt][nt][v] * 0.0625f;  // 1/sqrt(256)
  __syncthreads();

  // masked softmax per query row (lanes 0..19 each own one row)
  if (lane < 20) {
    float* row = Sc + lane * 32;
    float vals[20];
    float mx = -3.0e38f;
#pragma unroll
    for (int k = 0; k < 20; k++) {
      float v = (ids[k] != 0) ? row[k] : NEG_INF_F;
      vals[k] = v;
      mx = fmaxf(mx, v);
    }
    float sum = 0.f;
#pragma unroll
    for (int k = 0; k < 20; k++) {
      float e = expf(vals[k] - mx);
      vals[k] = e;
      sum += e;
    }
    float inv = 1.f / sum;
#pragma unroll
    for (int k = 0; k < 20; k++) row[k] = vals[k] * inv;
  }
  __syncthreads();

  if (lane < 20) {
    float t = 0.f;
#pragma unroll
    for (int q = 0; q < 20; q++) t += Sc[q * 32 + lane];
    wcol[lane] = t;
  }
  __syncthreads();

  // E_hat row = sum_k w[k] * X[k]
  float e[8];
#pragma unroll
  for (int t = 0; t < 8; t++) e[t] = 0.f;
  for (int k = 0; k < 20; k++) {
    float wk = wcol[k];
    const _Float16* xr = Xh + k * 256 + lane * 8;
#pragma unroll
    for (int t = 0; t < 8; t++) e[t] += wk * (float)xr[t];
  }
  float* out = Ehat + (size_t)s * 256 + lane * 8;
#pragma unroll
  for (int t = 0; t < 8; t++) out[t] = e[t];
}

// ---------------------------------------------------------------------------
// S = Ehat + positional encoding; u_short = Ehat[:, 49, :]
// ---------------------------------------------------------------------------
__global__ __launch_bounds__(256) void add_pe_kernel(
    const float* __restrict__ Ehat, float* __restrict__ S,
    float* __restrict__ u_short) {
  int m = blockIdx.x;
  int d = threadIdx.x;
  int pos = m % 50;
  int i = d >> 1;
  const float c = -9.21034037197618f / 256.0f;  // -ln(10000)/256
  float div = expf((float)(2 * i) * c);
  float ang = (float)pos * div;
  float pe = (d & 1) ? cosf(ang) : sinf(ang);
  float v = Ehat[(size_t)m * 256 + d];
  S[(size_t)m * 256 + d] = v + pe;
  if (pos == 49) u_short[(size_t)(m / 50) * 256 + d] = v;
}

// ---------------------------------------------------------------------------
// LayerNorm (ddof=1, eps added to sigma): one wave per row of 256.
// ---------------------------------------------------------------------------
__global__ __launch_bounds__(256) void ln_kernel(const float* __restrict__ X,
                                                 const float* __restrict__ g,
                                                 const float* __restrict__ bta,
                                                 float* __restrict__ Y) {
  int wid = (blockIdx.x * 256 + threadIdx.x) >> 5;
  int lane = threadIdx.x & 31;
  const float* x = X + (size_t)wid * 256 + lane * 8;
  float v[8];
  float s = 0.f;
#pragma unroll
  for (int t = 0; t < 8; t++) {
    v[t] = x[t];
    s += v[t];
  }
  float mean = wave_sum(s) * (1.0f / 256.0f);
  float q = 0.f;
#pragma unroll
  for (int t = 0; t < 8; t++) {
    float d = v[t] - mean;
    q += d * d;
  }
  float var = wave_sum(q) * (1.0f / 255.0f);
  float inv = 1.0f / (sqrtf(var) + 1e-6f);
  float* y = Y + (size_t)wid * 256 + lane * 8;
#pragma unroll
  for (int t = 0; t < 8; t++) {
    int d = lane * 8 + t;
    y[t] = g[d] * (v[t] - mean) * inv + bta[d];
  }
}

// ---------------------------------------------------------------------------
// Multi-head attention (INTER=50, dh=32): one wave per (batch, head).
// ---------------------------------------------------------------------------
__global__ __launch_bounds__(32) void attn_kernel(
    const float* __restrict__ QKV, const int* __restrict__ sessions,
    float* __restrict__ AttnOut) {
  __shared__ _Float16 Qh[64 * 32];
  __shared__ _Float16 Kh[64 * 32];
  __shared__ _Float16 Vh[64 * 32];
  __shared__ _Float16 Ph[64 * 64];
  __shared__ float Pf[64 * 64];
  __shared__ int valid[64];

  int g = blockIdx.x;
  int b = g >> 3;
  int h = g & 7;
  int lane = threadIdx.x;

  for (int i = lane; i < 64 * 32; i += 32) {
    Qh[i] = (_Float16)0.f;
    Kh[i] = (_Float16)0.f;
    Vh[i] = (_Float16)0.f;
  }
  for (int i = lane; i < 64 * 64; i += 32) Ph[i] = (_Float16)0.f;
  for (int i = lane; i < 64; i += 32)
    valid[i] = (i < 50) ? (sessions[(size_t)(b * 50 + i) * 20] != 0) : 0;
  __syncthreads();

  const float* base = QKV + (size_t)(b * 50) * 768 + h * 32;
  for (int r = 0; r < 50; r++) {
    const float* rp = base + (size_t)r * 768;
    Qh[r * 32 + lane] = (_Float16)rp[lane];
    Kh[r * 32 + lane] = (_Float16)rp[256 + lane];
    Vh[r * 32 + lane] = (_Float16)rp[512 + lane];
  }
  __syncthreads();

  // scores = Q K^T / sqrt(32)
  const float scale = 0.17677669529663687f;
#pragma unroll
  for (int mt = 0; mt < 4; mt++) {
    v16h a = a_frag(Qh, 32, mt * 16, 0, lane);
#pragma unroll
    for (int nt = 0; nt < 4; nt++) {
      v16h bf = bT_frag(Kh, 32, nt * 16, 0, lane);
      v8f c;
#pragma unroll
      for (int v = 0; v < 8; v++) c[v] = 0.f;
      c = wmma_f16(a, bf, c);
#pragma unroll
      for (int v = 0; v < 8; v++)
        Pf[(mt * 16 + v + 8 * (lane >> 4)) * 64 + nt * 16 + (lane & 15)] =
            c[v] * scale;
    }
  }
  __syncthreads();

  // masked softmax over keys for rows 0..49
  for (int r = lane; r < 50; r += 32) {
    float* row = Pf + r * 64;
    float mx = -3.0e38f;
    for (int k = 0; k < 50; k++) {
      float v = valid[k] ? row[k] : NEG_INF_F;
      row[k] = v;
      mx = fmaxf(mx, v);
    }
    float sum = 0.f;
    for (int k = 0; k < 50; k++) {
      float e = expf(row[k] - mx);
      row[k] = e;
      sum += e;
    }
    float inv = 1.f / sum;
    for (int k = 0; k < 50; k++) Ph[r * 64 + k] = (_Float16)(row[k] * inv);
  }
  __syncthreads();

  // O = P V  (K2 = 64 padded -> 2 k-tiles)
#pragma unroll
  for (int mt = 0; mt < 4; mt++) {
    v16h a0 = a_frag(Ph, 64, mt * 16, 0, lane);
    v16h a1 = a_frag(Ph, 64, mt * 16, 32, lane);
#pragma unroll
    for (int nt = 0; nt < 2; nt++) {
      v8f c;
#pragma unroll
      for (int v = 0; v < 8; v++) c[v] = 0.f;
      c = wmma_f16(a0, b_frag(Vh, 32, 0, nt * 16, lane), c);
      c = wmma_f16(a1, b_frag(Vh, 32, 32, nt * 16, lane), c);
#pragma unroll
      for (int v = 0; v < 8; v++) {
        int r = mt * 16 + v + 8 * (lane >> 4);
        if (r < 50)
          AttnOut[(size_t)(b * 50 + r) * 256 + h * 32 + nt * 16 + (lane & 15)] =
              c[v];
      }
    }
  }
}

// ---------------------------------------------------------------------------
// u_long = sum_i S1[b,i,:] + user_emb[batch_us[b]]
// ---------------------------------------------------------------------------
__global__ __launch_bounds__(256) void ulong_kernel(
    const float* __restrict__ S1, const float* __restrict__ user_emb,
    const int* __restrict__ batch_us, float* __restrict__ u_long) {
  int b = blockIdx.x;
  int d = threadIdx.x;
  float t = 0.f;
  for (int i = 0; i < 50; i++) t += S1[(size_t)(b * 50 + i) * 256 + d];
  t += user_emb[(size_t)batch_us[b] * 256 + d];
  u_long[(size_t)b * 256 + d] = t;
}

// ---------------------------------------------------------------------------
// Gate: Tg = sigmoid(ul@Wl + us@Ws + t@Wt + bias); U = Tg*us + (1-Tg)*ul
// ---------------------------------------------------------------------------
__global__ __launch_bounds__(256) void gate_kernel(
    const float* __restrict__ u_long, const float* __restrict__ u_short,
    const float* __restrict__ time_emb, const int* __restrict__ batch_td,
    const float* __restrict__ Wl, const float* __restrict__ Ws,
    const float* __restrict__ Wt, const float* __restrict__ gbias,
    float* __restrict__ U) {
  __shared__ float sul[256], sus[256], ste[256];
  int b = blockIdx.x;
  int j = threadIdx.x;
  sul[j] = u_long[(size_t)b * 256 + j];
  sus[j] = u_short[(size_t)b * 256 + j];
  ste[j] = time_emb[(size_t)batch_td[b] * 256 + j];
  __syncthreads();
  float z = gbias[j];
  for (int d = 0; d < 256; d++) {
    z += sul[d] * Wl[(size_t)d * 256 + j];
    z += sus[d] * Ws[(size_t)d * 256 + j];
    z += ste[d] * Wt[(size_t)d * 256 + j];
  }
  float tg = 1.0f / (1.0f + expf(-z));
  U[(size_t)b * 256 + j] = tg * sus[j] + (1.0f - tg) * sul[j];
}

// ---------------------------------------------------------------------------
// Scoring: out[b][c] = dot(item_emb[idx], U[b]); wave per candidate (gather bound)
// ---------------------------------------------------------------------------
__global__ __launch_bounds__(256) void score_kernel(
    const float* __restrict__ item_emb, const int* __restrict__ cand,
    const float* __restrict__ U, float* __restrict__ out) {
  __shared__ float Ush[256];
  int b = blockIdx.x;
  Ush[threadIdx.x] = U[(size_t)b * 256 + threadIdx.x];
  __syncthreads();
  int wid = threadIdx.x >> 5;
  int lane = threadIdx.x & 31;
  const float* u = Ush + lane * 8;
  for (int c = wid; c < 2048; c += 8) {
    int idx = cand[(size_t)b * 2048 + c];
    const float4* p = (const float4*)(item_emb + (size_t)idx * 256 + lane * 8);
    float4 x0 = p[0], x1 = p[1];
    float t = x0.x * u[0] + x0.y * u[1] + x0.z * u[2] + x0.w * u[3] +
              x1.x * u[4] + x1.y * u[5] + x1.z * u[6] + x1.w * u[7];
    t = wave_sum(t);
    if (lane == 0) out[(size_t)b * 2048 + c] = t;
  }
}

// ---------------------------------------------------------------------------
// Host-side launcher
// ---------------------------------------------------------------------------
extern "C" void kernel_launch(void* const* d_in, const int* in_sizes, int n_in,
                              void* d_out, int out_size, void* d_ws,
                              size_t ws_size, hipStream_t stream) {
  (void)in_sizes; (void)n_in; (void)out_size; (void)ws_size;

  const float* item_emb = (const float*)d_in[0];
  const float* user_emb = (const float*)d_in[1];
  const float* time_emb = (const float*)d_in[2];
  const float* Wq = (const float*)d_in[3];
  const float* bq = (const float*)d_in[4];
  const float* Wk = (const float*)d_in[5];
  const float* bk = (const float*)d_in[6];
  const float* Wv = (const float*)d_in[7];
  const float* bv = (const float*)d_in[8];
  const float* Wo = (const float*)d_in[9];
  const float* bo = (const float*)d_in[10];
  const float* ln1_g = (const float*)d_in[11];
  const float* ln1_b = (const float*)d_in[12];
  const float* W1 = (const float*)d_in[13];
  const float* b1 = (const float*)d_in[14];
  const float* W2 = (const float*)d_in[15];
  const float* b2 = (const float*)d_in[16];
  const float* ln2_g = (const float*)d_in[17];
  const float* ln2_b = (const float*)d_in[18];
  const float* fn_g = (const float*)d_in[19];
  const float* fn_b = (const float*)d_in[20];
  const float* gate_Wl = (const float*)d_in[21];
  const float* gate_Ws = (const float*)d_in[22];
  const float* gate_Wt = (const float*)d_in[23];
  const float* gate_bias = (const float*)d_in[24];
  const int* batch_us = (const int*)d_in[25];
  const int* batch_sessions = (const int*)d_in[26];
  const int* batch_td = (const int*)d_in[27];
  const int* batch_pred = (const int*)d_in[28];
  float* out = (float*)d_out;

  // ---- workspace layout (256B aligned), with lifetime-based aliasing ----
  char* ws = (char*)d_ws;
  size_t off = 0;
  auto take = [&](size_t bytes) -> char* {
    char* p = ws + off;
    off += (bytes + 255) & ~(size_t)255;
    return p;
  };
  _Float16* WQKVpk = (_Float16*)take(48 * 8 * 1024);   // K=256,N=768
  float* bqkv = (float*)take(768 * 4);
  _Float16* WOpk = (_Float16*)take(16 * 8 * 1024);     // K=256,N=256
  _Float16* W1pk = (_Float16*)take(64 * 8 * 1024);     // K=256,N=1024
  _Float16* W2pk = (_Float16*)take(16 * 32 * 1024);    // K=1024,N=256
  float* Ehat = (float*)take((size_t)12800 * 256 * 4);
  float* S = (float*)take((size_t)12800 * 256 * 4);
  float* Yf = (float*)take((size_t)12800 * 256 * 4);   // LN out; alias AttnOut
  float* AttnOut = Yf;
  char* pkbuf = take((size_t)800 * 32 * 1024);         // Ypk/AOpk/Hpk (disjoint lifetimes)
  _Float16* Ypk = (_Float16*)pkbuf;
  _Float16* AOpk = (_Float16*)pkbuf;
  _Float16* Hpk = (_Float16*)pkbuf;
  char* big = take((size_t)12800 * 1024 * 4);          // QKV (39MB) / Hf (52MB)
  float* QKV = (float*)big;
  float* Hf = (float*)big;
  float* u_short = (float*)take(256 * 256 * 4);
  float* u_long = (float*)take(256 * 256 * 4);
  float* Ugate = (float*)take(256 * 256 * 4);

  // ---- pack weights into B-fragment layout (done every call; cheap) ----
  pack_b_kernel<<<32, 128, 0, stream>>>(Wq, WQKVpk, 8, 256, 0);
  pack_b_kernel<<<32, 128, 0, stream>>>(Wk, WQKVpk, 8, 256, 16);
  pack_b_kernel<<<32, 128, 0, stream>>>(Wv, WQKVpk, 8, 256, 32);
  pack_b_kernel<<<32, 128, 0, stream>>>(Wo, WOpk, 8, 256, 0);
  pack_b_kernel<<<128, 128, 0, stream>>>(W1, W1pk, 8, 1024, 0);
  pack_b_kernel<<<128, 128, 0, stream>>>(W2, W2pk, 32, 256, 0);
  concat_bias_kernel<<<3, 256, 0, stream>>>(bq, bk, bv, bqkv);

  // ---- intra-session attention + positional encoding ----
  intra_kernel<<<12800, 32, 0, stream>>>(item_emb, batch_sessions, Ehat);
  add_pe_kernel<<<12800, 256, 0, stream>>>(Ehat, S, u_short);

  // ---- 2 transformer blocks (same weights per reference) ----
  for (int blk = 0; blk < 2; blk++) {
    ln_kernel<<<1600, 256, 0, stream>>>(S, ln1_g, ln1_b, Yf);
    pack_a_kernel<<<1600, 128, 0, stream>>>(Yf, Ypk, 8, 256);
    gemm_kernel<8, false, false><<<2400, 128, 0, stream>>>(
        Ypk, WQKVpk, bqkv, nullptr, QKV, 12, 768);
    attn_kernel<<<2048, 32, 0, stream>>>(QKV, batch_sessions, AttnOut);
    pack_a_kernel<<<1600, 128, 0, stream>>>(AttnOut, AOpk, 8, 256);
    gemm_kernel<8, false, true><<<800, 128, 0, stream>>>(
        AOpk, WOpk, bo, S, S, 4, 256);
    ln_kernel<<<1600, 256, 0, stream>>>(S, ln2_g, ln2_b, Yf);
    pack_a_kernel<<<1600, 128, 0, stream>>>(Yf, Ypk, 8, 256);
    gemm_kernel<8, true, false><<<3200, 128, 0, stream>>>(
        Ypk, W1pk, b1, nullptr, Hf, 16, 1024);
    pack_a_kernel<<<6400, 128, 0, stream>>>(Hf, Hpk, 32, 1024);
    gemm_kernel<32, false, true><<<800, 128, 0, stream>>>(
        Hpk, W2pk, b2, S, S, 4, 256);
  }

  // ---- final norm, gate, scoring ----
  ln_kernel<<<1600, 256, 0, stream>>>(S, fn_g, fn_b, Yf);
  ulong_kernel<<<256, 256, 0, stream>>>(Yf, user_emb, batch_us, u_long);
  gate_kernel<<<256, 256, 0, stream>>>(u_long, u_short, time_emb, batch_td,
                                       gate_Wl, gate_Ws, gate_Wt, gate_bias,
                                       Ugate);
  score_kernel<<<256, 256, 0, stream>>>(item_emb, batch_pred, Ugate, out);
}
